// QKNet_3143916060954
// MI455X (gfx1250) — matrix operand order repaired and checked by reading, not compiled, not run on hardware
//
#include <hip/hip_runtime.h>

// ---------------------------------------------------------------------------
// Types for CDNA5 WMMA (wave32): v_wmma_f32_16x16x32_bf16
// ---------------------------------------------------------------------------
typedef __attribute__((ext_vector_type(16))) __bf16 bf16x16;
typedef __attribute__((ext_vector_type(8)))  __bf16 bf16x8;
typedef __attribute__((ext_vector_type(8)))  float  f32x8;

#define WMMA_BF16(a, b, c) \
  __builtin_amdgcn_wmma_f32_16x16x32_bf16(false, (a), false, (b), (short)0, (c), false, false)

// Problem constants (from the reference)
static constexpr int BATCH = 64;
static constexpr int C1N   = 64;
static constexpr int C2N   = 128;
static constexpr int KCEN  = 512;
static constexpr int D0    = 1024;   // 32*32
static constexpr int D1    = 256;    // 16*16
static constexpr int K2IM  = 1600;   // C1 * 25 (conv2 implicit-GEMM K)
static constexpr int FC1K  = 32768;  // C2 * 256
static constexpr int FC1N  = 1024;
static constexpr int FC1SPLIT = 8;   // K-split factor for fc1 (bandwidth bound)

// ---------------------------------------------------------------------------
// Fragment loaders.
// A (16x32 bf16, M rows x K): lane l (l<16): M=l, halves = K{c0..c0+7, c0+16..c0+23}
//                             lane l>=16:   M=l-16, halves shifted by +8.
// B (32x16 bf16, K x N), loaded from a row-major B^T (N x K) f32 buffer:
//   lane l: N = l&15, halves = K{c0..c0+7, c0+16..c0+23} shifted by +8 for l>=16 (same as A).
// C/D (16x16 f32): lane l, vgpr j -> (M = j + (l>>4)*8, N = l&15).
// ---------------------------------------------------------------------------
__device__ inline bf16x16 loadA_bf16(const __bf16* base, long long rowStride,
                                     int c0, int lane) {
  const int m  = lane & 15;
  const int hi = lane >> 4;
  const __bf16* r = base + (long long)m * rowStride + c0 + hi * 8;
  bf16x8 lo8 = *(const bf16x8*)(r);
  bf16x8 hi8 = *(const bf16x8*)(r + 16);
  bf16x16 a;
#pragma unroll
  for (int i = 0; i < 8; ++i) { a[i] = lo8[i]; a[8 + i] = hi8[i]; }
  return a;
}

__device__ inline bf16x16 loadBT_f32(const float* bt, long long rowStride,
                                     int c0, int lane) {
  const int n  = lane & 15;
  const int hi = lane >> 4;
  const float* r = bt + (long long)n * rowStride + c0 + hi * 8;
  bf16x16 b;
#pragma unroll
  for (int i = 0; i < 16; ++i) b[i] = (__bf16)r[i < 8 ? i : i + 8];
  return b;
}

// ---------------------------------------------------------------------------
// K0: conv1(5x5,pad2) + bias + ReLU + maxpool2 + row L2-normalize -> bf16 V0
// One block per (b, c1); 256 threads, 4 pooled outputs each (D0 = 1024).
// ---------------------------------------------------------------------------
__global__ void conv1_pool_norm(const float* __restrict__ X,
                                const float* __restrict__ W1,
                                const float* __restrict__ B1,
                                __bf16* __restrict__ V0) {
  __shared__ float wsh[75];
  __shared__ float red[256];
  const int blk = blockIdx.x;
  const int b   = blk >> 6;     // / C1N
  const int c1  = blk & 63;
  const int tid = threadIdx.x;
  if (tid < 75) wsh[tid] = W1[c1 * 75 + tid];
  __syncthreads();
  const float bias = B1[c1];
  const float* Xb = X + (long long)b * 3 * 64 * 64;

  float outv[4];
  float ss = 0.f;
#pragma unroll
  for (int i = 0; i < 4; ++i) {
    const int p  = tid + i * 256;       // pooled pixel 0..1023
    const int py = p >> 5, px = p & 31;
    float mx = -3.0e38f;
#pragma unroll
    for (int sy = 0; sy < 2; ++sy)
#pragma unroll
      for (int sx = 0; sx < 2; ++sx) {
        const int y = py * 2 + sy, x = px * 2 + sx;
        float acc = bias;
        for (int ci = 0; ci < 3; ++ci)
#pragma unroll
          for (int dy = 0; dy < 5; ++dy) {
            const int yy = y + dy - 2;
            if ((unsigned)yy >= 64u) continue;
#pragma unroll
            for (int dx = 0; dx < 5; ++dx) {
              const int xx = x + dx - 2;
              if ((unsigned)xx >= 64u) continue;
              acc += Xb[ci * 4096 + yy * 64 + xx] * wsh[ci * 25 + dy * 5 + dx];
            }
          }
        mx = fmaxf(mx, acc);
      }
    mx = fmaxf(mx, 0.f);                // ReLU (commutes with max)
    outv[i] = mx;
    ss += mx * mx;
  }
  red[tid] = ss;
  __syncthreads();
  for (int s = 128; s > 0; s >>= 1) {
    if (tid < s) red[tid] += red[tid + s];
    __syncthreads();
  }
  const float scale = 1.f / fmaxf(sqrtf(red[0]), 1e-12f);
  __bf16* out = V0 + ((long long)b * C1N + c1) * D0;
#pragma unroll
  for (int i = 0; i < 4; ++i) out[tid + i * 256] = (__bf16)(outv[i] * scale);
}

// ---------------------------------------------------------------------------
// K1: 0.5 * ||center||^2 per (c,k) row.  One wave per row, 4 waves/block.
// ---------------------------------------------------------------------------
__global__ void cen_halfnorm(const float* __restrict__ cen,
                             float* __restrict__ nrm, int D) {
  const int row  = blockIdx.x * 4 + (threadIdx.x >> 5);
  const int lane = threadIdx.x & 31;
  const float* r = cen + (long long)row * D;
  float s = 0.f;
  for (int d = lane; d < D; d += 32) { const float v = r[d]; s += v * v; }
#pragma unroll
  for (int o = 16; o; o >>= 1) s += __shfl_xor(s, o, 32);
  if (lane == 0) nrm[row] = 0.5f * s;
}

// ---------------------------------------------------------------------------
// K2: per-channel scores  S[c][b][k] = sum_d V[b][c][d] * cen[c][k][d]
// WMMA bf16: M=64 (batch, 4 tiles), N=512 (centers), K=D.
// grid = (C, 8); block = 128 (4 waves); wave -> one 16-col N-tile.
// ---------------------------------------------------------------------------
__global__ void knn_scores_wmma(const __bf16* __restrict__ V,
                                const float* __restrict__ cen,
                                float* __restrict__ S, int C, int D) {
  const int c     = blockIdx.x;
  const int wave  = threadIdx.x >> 5;
  const int lane  = threadIdx.x & 31;
  const int ntile = blockIdx.y * 4 + wave;            // 0..31 (512 cols)
  const long long rowStrideA = (long long)C * D;      // batch stride in V
  const float* BT = cen + ((long long)c * KCEN + (long long)ntile * 16) * D;
  const __bf16* A0 = V + (long long)c * D;            // batch 0, channel c

  f32x8 acc[4] = {};
  for (int c0 = 0; c0 < D; c0 += 32) {
    const bf16x16 bf = loadBT_f32(BT, D, c0, lane);
#pragma unroll
    for (int mt = 0; mt < 4; ++mt) {
      const bf16x16 af =
          loadA_bf16(A0 + (long long)(mt * 16) * rowStrideA, rowStrideA, c0, lane);
      acc[mt] = WMMA_BF16(af, bf, acc[mt]);
    }
  }
  const int n     = ntile * 16 + (lane & 15);
  const int mbase = (lane >> 4) * 8;
#pragma unroll
  for (int mt = 0; mt < 4; ++mt)
#pragma unroll
    for (int j = 0; j < 8; ++j) {
      const int m = mt * 16 + mbase + j;
      S[((long long)c * BATCH + m) * KCEN + n] = acc[mt][j];
    }
}

// ---------------------------------------------------------------------------
// K3: argmax_k (S - 0.5||cen||^2)  (== argmin distance, first-index ties),
// then gather the winning center row as bf16 into Q[b][c][:].
// One wave per (c,b) pair; 4 waves/block.
// ---------------------------------------------------------------------------
__global__ void argmin_gather(const float* __restrict__ S,
                              const float* __restrict__ nrm,
                              const float* __restrict__ cen,
                              __bf16* __restrict__ Q, int C, int D) {
  const int pair = blockIdx.x * 4 + (threadIdx.x >> 5);
  const int lane = threadIdx.x & 31;
  const int c = pair >> 6;          // pair = c*64 + b
  const int b = pair & 63;
  const float* s  = S + (long long)pair * KCEN;
  const float* nr = nrm + (long long)c * KCEN;
  float best = -3.0e38f;
  int   bi   = 0;
  for (int k = lane; k < KCEN; k += 32) {
    const float v = s[k] - nr[k];
    if (v > best) { best = v; bi = k; }
  }
#pragma unroll
  for (int o = 16; o; o >>= 1) {
    const float ob  = __shfl_xor(best, o, 32);
    const int   obi = __shfl_xor(bi, o, 32);
    if (ob > best || (ob == best && obi < bi)) { best = ob; bi = obi; }
  }
  const float* q = cen + ((long long)c * KCEN + bi) * D;
  __bf16* out = Q + ((long long)b * C + c) * D;
  for (int d = lane; d < D; d += 32) out[d] = (__bf16)q[d];
}

// ---------------------------------------------------------------------------
// K4: conv2 (5x5, pad2, 64->128) as WMMA implicit GEMM + bias + ReLU.
// M = pixels (64/block, 4 tiles), N = 128 channels (8 waves x 1 tile),
// K = 1600 (50 chunks of 32).  A tile im2col-staged in LDS each chunk.
// grid = B*16; block = 256 (8 waves).
// ---------------------------------------------------------------------------
__global__ void conv2_wmma(const __bf16* __restrict__ Q0,
                           const float* __restrict__ W2,
                           const float* __restrict__ B2,
                           float* __restrict__ H2) {
  __shared__ __bf16 tileA[64 * 32];
  const int blk     = blockIdx.x;
  const int b       = blk >> 4;
  const int pixBase = (blk & 15) * 64;
  const int wave    = threadIdx.x >> 5;
  const int lane    = threadIdx.x & 31;
  const __bf16* In  = Q0 + (long long)b * C1N * 1024;

  f32x8 acc[4] = {};
  for (int c0 = 0; c0 < K2IM; c0 += 32) {
    __syncthreads();
    // cooperative im2col gather into LDS: 64 pixels x 32 k-values
    for (int t = threadIdx.x; t < 64 * 32; t += 256) {
      const int m  = t >> 5;
      const int k  = c0 + (t & 31);
      const int ci = k / 25;
      const int r  = k - ci * 25;
      const int dy = r / 5 - 2;
      const int dx = r - (r / 5) * 5 - 2;
      const int p  = pixBase + m;
      const int y  = (p >> 5) + dy;
      const int x  = (p & 31) + dx;
      __bf16 v = (__bf16)0.f;
      if ((unsigned)y < 32u && (unsigned)x < 32u)
        v = In[(long long)ci * 1024 + y * 32 + x];
      tileA[t] = v;
    }
    __syncthreads();
    const bf16x16 bf = loadBT_f32(W2 + (long long)(wave * 16) * K2IM, K2IM, c0, lane);
#pragma unroll
    for (int mt = 0; mt < 4; ++mt) {
      const bf16x16 af = loadA_bf16(tileA + mt * 16 * 32, 32, 0, lane);
      acc[mt] = WMMA_BF16(af, bf, acc[mt]);
    }
  }
  const int co    = wave * 16 + (lane & 15);
  const float bb  = B2[co];
  const int mbase = (lane >> 4) * 8;
#pragma unroll
  for (int mt = 0; mt < 4; ++mt)
#pragma unroll
    for (int j = 0; j < 8; ++j) {
      const int p = pixBase + mt * 16 + mbase + j;
      const float v = fmaxf(acc[mt][j] + bb, 0.f);
      H2[((long long)b * C2N + co) * 1024 + p] = v;
    }
}

// ---------------------------------------------------------------------------
// K5: maxpool2 + L2-normalize for H2 -> bf16 V1. Block per (b,c2); 64 thr.
// ---------------------------------------------------------------------------
__global__ void pool_norm(const float* __restrict__ Hin,
                          __bf16* __restrict__ Vout, int C) {
  __shared__ float red[64];
  const int blk = blockIdx.x;
  const int b = blk / C, c = blk % C;
  const float* in = Hin + ((long long)b * C + c) * 1024;
  const int tid = threadIdx.x;
  float vals[4];
  float ss = 0.f;
#pragma unroll
  for (int i = 0; i < 4; ++i) {
    const int p  = tid + i * 64;        // 0..255
    const int py = p >> 4, px = p & 15;
    const float* r = in + (py * 2) * 32 + px * 2;
    const float m = fmaxf(fmaxf(r[0], r[1]), fmaxf(r[32], r[33]));
    vals[i] = m;
    ss += m * m;
  }
  red[tid] = ss;
  __syncthreads();
  for (int s = 32; s > 0; s >>= 1) {
    if (tid < s) red[tid] += red[tid + s];
    __syncthreads();
  }
  const float scale = 1.f / fmaxf(sqrtf(red[0]), 1e-12f);
  __bf16* out = Vout + ((long long)b * C + c) * D1;
#pragma unroll
  for (int i = 0; i < 4; ++i) out[tid + i * 64] = (__bf16)(vals[i] * scale);
}

// ---------------------------------------------------------------------------
// K6a: zero the fc1 f32 partial-sum buffer (64x1024).
// ---------------------------------------------------------------------------
__global__ void zero_f32(float* __restrict__ p, int n) {
  const int t = blockIdx.x * blockDim.x + threadIdx.x;
  if (t < n) p[t] = 0.f;
}

// ---------------------------------------------------------------------------
// K6b: fc1 partial GEMM, K split 8 ways across gridDim.y for HBM bandwidth.
// Each block: 4 waves x one 16-col N-tile over K range [y*4096, (y+1)*4096).
// Partials accumulated with global f32 atomics into Yf (64x1024).
// grid = (16, 8); block = 128.
// ---------------------------------------------------------------------------
__global__ void fc1_wmma_split(const __bf16* __restrict__ Xb,
                               const float* __restrict__ W,
                               float* __restrict__ Yf) {
  const int wave  = threadIdx.x >> 5;
  const int lane  = threadIdx.x & 31;
  const int ntile = blockIdx.x * 4 + wave;            // 0..63
  const int kspan = FC1K / FC1SPLIT;                  // 4096
  const int k0    = blockIdx.y * kspan;
  const float* BT = W + (long long)(ntile * 16) * FC1K;

  f32x8 acc[4] = {};
  for (int c0 = k0; c0 < k0 + kspan; c0 += 32) {
    const bf16x16 bf = loadBT_f32(BT, FC1K, c0, lane);
#pragma unroll
    for (int mt = 0; mt < 4; ++mt) {
      const bf16x16 af =
          loadA_bf16(Xb + (long long)(mt * 16) * FC1K, FC1K, c0, lane);
      acc[mt] = WMMA_BF16(af, bf, acc[mt]);
    }
  }
  const int n     = ntile * 16 + (lane & 15);
  const int mbase = (lane >> 4) * 8;
#pragma unroll
  for (int mt = 0; mt < 4; ++mt)
#pragma unroll
    for (int j = 0; j < 8; ++j) {
      const int m = mt * 16 + mbase + j;
      atomicAdd(&Yf[(long long)m * FC1N + n], acc[mt][j]);
    }
}

// ---------------------------------------------------------------------------
// K6c: fc1 epilogue: bias + ReLU + convert to bf16.
// ---------------------------------------------------------------------------
__global__ void fc1_finish(const float* __restrict__ Yf,
                           const float* __restrict__ Bb,
                           __bf16* __restrict__ Y) {
  const int t = blockIdx.x * blockDim.x + threadIdx.x;   // 0..65535
  const int n = t & (FC1N - 1);
  const float v = fmaxf(Yf[t] + Bb[n], 0.f);
  Y[t] = (__bf16)v;
}

// ---------------------------------------------------------------------------
// K7: fc2 (64x1024) x (1024x10) + b -> d_out (64x10 f32). Tiny; scalar.
// ---------------------------------------------------------------------------
__global__ void fc2_kernel(const __bf16* __restrict__ H3,
                           const float* __restrict__ W,
                           const float* __restrict__ Bb,
                           float* __restrict__ Out) {
  const int t = blockIdx.x * blockDim.x + threadIdx.x;
  if (t >= BATCH * 10) return;
  const int b = t / 10, cls = t - b * 10;
  const __bf16* h = H3 + (long long)b * FC1N;
  const float*  w = W + (long long)cls * FC1N;
  float acc = Bb[cls];
  for (int j = 0; j < FC1N; ++j) acc += (float)h[j] * w[j];
  Out[t] = acc;
}

// ---------------------------------------------------------------------------
// Host-side launch
// ---------------------------------------------------------------------------
extern "C" void kernel_launch(void* const* d_in, const int* in_sizes, int n_in,
                              void* d_out, int out_size, void* d_ws, size_t ws_size,
                              hipStream_t stream) {
  const float* x    = (const float*)d_in[0];
  const float* w1   = (const float*)d_in[1];
  const float* b1   = (const float*)d_in[2];
  const float* w2   = (const float*)d_in[3];
  const float* b2   = (const float*)d_in[4];
  const float* fw1  = (const float*)d_in[5];
  const float* fb1  = (const float*)d_in[6];
  const float* fw2  = (const float*)d_in[7];
  const float* fb2  = (const float*)d_in[8];
  const float* cen0 = (const float*)d_in[9];
  const float* cen1 = (const float*)d_in[10];
  float* out = (float*)d_out;

  // Workspace carve-up (all offsets 256B-aligned by construction)
  char* ws = (char*)d_ws;
  size_t off = 0;
  __bf16* V0  = (__bf16*)(ws + off); off += (size_t)BATCH * C1N * D0 * 2;   // 8 MB
  __bf16* Q0  = (__bf16*)(ws + off); off += (size_t)BATCH * C1N * D0 * 2;   // 8 MB
  __bf16* V1  = (__bf16*)(ws + off); off += (size_t)BATCH * C2N * D1 * 2;   // 4 MB
  __bf16* Q1  = (__bf16*)(ws + off); off += (size_t)BATCH * C2N * D1 * 2;   // 4 MB
  float*  H2  = (float*)(ws + off);  off += (size_t)BATCH * C2N * 1024 * 4; // 32 MB
  float*  N0  = (float*)(ws + off);  off += (size_t)C1N * KCEN * 4;
  float*  N1  = (float*)(ws + off);  off += (size_t)C2N * KCEN * 4;
  float*  S0  = (float*)(ws + off);  off += (size_t)C1N * BATCH * KCEN * 4; // 8 MB
  float*  S1  = (float*)(ws + off);  off += (size_t)C2N * BATCH * KCEN * 4; // 16 MB
  float*  H3f = (float*)(ws + off);  off += (size_t)BATCH * FC1N * 4;       // f32 partials
  __bf16* H3  = (__bf16*)(ws + off); off += (size_t)BATCH * FC1N * 2;

  // Stage 1: conv1 + pool + normalize
  conv1_pool_norm<<<BATCH * C1N, 256, 0, stream>>>(x, w1, b1, V0);
  // KNN layer 0
  cen_halfnorm<<<(C1N * KCEN) / 4, 128, 0, stream>>>(cen0, N0, D0);
  knn_scores_wmma<<<dim3(C1N, 8), 128, 0, stream>>>(V0, cen0, S0, C1N, D0);
  argmin_gather<<<(C1N * BATCH) / 4, 128, 0, stream>>>(S0, N0, cen0, Q0, C1N, D0);
  // Stage 2: conv2 (WMMA implicit GEMM) + pool + normalize
  conv2_wmma<<<BATCH * 16, 256, 0, stream>>>(Q0, w2, b2, H2);
  pool_norm<<<BATCH * C2N, 64, 0, stream>>>(H2, V1, C2N);
  // KNN layer 1
  cen_halfnorm<<<(C2N * KCEN) / 4, 128, 0, stream>>>(cen1, N1, D1);
  knn_scores_wmma<<<dim3(C2N, 8), 128, 0, stream>>>(V1, cen1, S1, C2N, D1);
  argmin_gather<<<(C2N * BATCH) / 4, 128, 0, stream>>>(S1, N1, cen1, Q1, C2N, D1);
  // Classifier head: fc1 with 8-way K split (bandwidth-bound -> more WGPs)
  zero_f32<<<(BATCH * FC1N + 255) / 256, 256, 0, stream>>>(H3f, BATCH * FC1N);
  fc1_wmma_split<<<dim3(16, FC1SPLIT), 128, 0, stream>>>(Q1, fw1, H3f);
  fc1_finish<<<(BATCH * FC1N) / 256, 256, 0, stream>>>(H3f, fb1, H3);
  fc2_kernel<<<10, 64, 0, stream>>>(H3, fw2, fb2, out);

  (void)in_sizes; (void)n_in; (void)out_size; (void)ws_size;
}
